// Attention_65962107732002
// MI455X (gfx1250) — matrix operand order repaired and checked by reading, not compiled, not run on hardware
//
#include <hip/hip_runtime.h>

#define B_  4
#define S_  1024
#define H_  12
#define DM_ 768
#define DH_ 64
#define IGNORE_ (-100000.0f)

typedef __attribute__((ext_vector_type(16))) _Float16 v16h;
typedef __attribute__((ext_vector_type(8)))  _Float16 v8h;
typedef __attribute__((ext_vector_type(4)))  _Float16 h4;
typedef __attribute__((ext_vector_type(8)))  float    v8f;

// Build a 16x32 f16 A-fragment from two contiguous 8-element (16B) chunks.
// As called here: lane l<16 holds M=l, K={0..7} in elems 0-7 and K={8..15} in
// elems 8-15; lane l>=16 holds K={16..23} and K={24..31}.
__device__ __forceinline__ v16h mk_frag(const _Float16* p_lo, const _Float16* p_hi) {
  v8h lo = *(const v8h*)p_lo;
  v8h hi = *(const v8h*)p_hi;
  v16h r;
#pragma unroll
  for (int i = 0; i < 8; ++i) { r[i] = (_Float16)lo[i]; r[i + 8] = (_Float16)hi[i]; }
  return r;
}

__device__ __forceinline__ v8f wmma16(v16h a, v16h b, v8f c) {
  return __builtin_amdgcn_wmma_f32_16x16x32_f16(false, a, false, b, (short)0, c,
                                                false, false);
}

// ---------------------------------------------------------------------------
// Kernel 1: fused Q/K/V projection.  grid = (B*H, S/64), block = 128 (4 waves).
// Stages each x tile ONCE and computes all three projections from it
// (12 WMMAs per 32-wide k-step), so x is read from HBM exactly once.
// Writes Q/K row-major [bh][s][dh] and V transposed [bh][dh][s], all f16,
// via LDS transpose staging so every global store is a 16B b128.
// ---------------------------------------------------------------------------
__global__ __launch_bounds__(128) void qkv_kernel(
    const float* __restrict__ x,
    const float* __restrict__ Wq, const float* __restrict__ bq,
    const float* __restrict__ Wk, const float* __restrict__ bk,
    const float* __restrict__ Wv, const float* __restrict__ bv,
    _Float16* __restrict__ Qh, _Float16* __restrict__ Kh,
    _Float16* __restrict__ Vt) {
  // lds[0..2047]  : x tile   [row 64][k 32]
  // lds[2048*(p+1)..]: W tile p, transposed [n 64][k 32]
  __shared__ __align__(32) _Float16 lds[4 * 2048];   // 16 KB
  _Float16* ldsx = lds;

  const int bh = blockIdx.x;
  const int b  = bh / H_;
  const int h  = bh % H_;
  const int s0 = blockIdx.y * 64;

  const int tid   = threadIdx.x;
  const int lane  = tid & 31;
  const int w     = tid >> 5;        // wave id 0..3
  const int half_ = lane >> 4;       // 0 or 1
  const int l16   = lane & 15;

  const float* Wp[3]    = {Wq + (size_t)h * DM_ * DH_,
                           Wk + (size_t)h * DM_ * DH_,
                           Wv + (size_t)h * DM_ * DH_};
  const float* biasp[3] = {bq + (size_t)h * DH_,
                           bk + (size_t)h * DH_,
                           bv + (size_t)h * DH_};

  const float* xbase = x + (((size_t)b * S_ + s0) * H_ + h) * DM_;

  v8f acc[3][4] = {};

  for (int kk = 0; kk < DM_; kk += 32) {
    // ---- stage x tile 64x32: 4 float4 (b128) loads per thread
#pragma unroll
    for (int i = 0; i < 4; ++i) {
      const int e = tid + i * 128;                 // float4 chunk 0..511
      const int r = e >> 3, c = (e & 7) * 4;
      const float4 v = *(const float4*)(xbase + (size_t)r * (H_ * DM_) + kk + c);
      h4 hh = {(_Float16)v.x, (_Float16)v.y, (_Float16)v.z, (_Float16)v.w};
      *(h4*)(ldsx + r * 32 + c) = hh;
    }
    // ---- stage all 3 W tiles 32x64 transposed into [n][k]
#pragma unroll
    for (int p = 0; p < 3; ++p) {
      _Float16* ldsw = lds + (p + 1) * 2048;
#pragma unroll
      for (int i = 0; i < 4; ++i) {
        const int e = tid + i * 128;               // float4 chunk 0..511
        const int kr = e >> 4, n = (e & 15) * 4;
        const float4 v = *(const float4*)(Wp[p] + (size_t)(kk + kr) * DH_ + n);
        ldsw[(n + 0) * 32 + kr] = (_Float16)v.x;
        ldsw[(n + 1) * 32 + kr] = (_Float16)v.y;
        ldsw[(n + 2) * 32 + kr] = (_Float16)v.z;
        ldsw[(n + 3) * 32 + kr] = (_Float16)v.w;
      }
    }
    // ---- prefetch next x k-tile (global_prefetch_b8 path)
    if (kk + 32 < DM_) {
      const int r = tid >> 3, c = (tid & 7) * 4;
      __builtin_prefetch(xbase + (size_t)r * (H_ * DM_) + kk + 32 + c, 0, 3);
    }
    __syncthreads();

    const int arow = w * 16 + l16;
    const v16h a = mk_frag(ldsx + arow * 32 + half_ * 16,
                           ldsx + arow * 32 + half_ * 16 + 8);
#pragma unroll
    for (int p = 0; p < 3; ++p) {
      const _Float16* ldsw = lds + (p + 1) * 2048;
#pragma unroll
      for (int t = 0; t < 4; ++t) {
        const int n = t * 16 + l16;
        // B 32x16: lane holds N=l16, K = half_*16 + j (one 32B LDS load)
        const v16h bf = *(const v16h*)(ldsw + n * 32 + half_ * 16);
        acc[p][t] = wmma16(a, bf, acc[p][t]);
      }
    }
    __syncthreads();
  }

  // ---- epilogue: +bias, LDS transpose staging, b128 global stores
#pragma unroll
  for (int p = 0; p < 3; ++p) {
    float bb[4];
#pragma unroll
    for (int t = 0; t < 4; ++t) bb[t] = biasp[p][t * 16 + l16];
#pragma unroll
    for (int t = 0; t < 4; ++t) {
      const int n = t * 16 + l16;
#pragma unroll
      for (int r = 0; r < 8; ++r) {
        const int m = w * 16 + half_ * 8 + r;
        const float val = acc[p][t][r] + bb[t];
        if (p == 2) lds[n * 64 + m] = (_Float16)val;   // V as [d][s]
        else        lds[m * 64 + n] = (_Float16)val;   // Q/K as [s][d]
      }
    }
    __syncthreads();
#pragma unroll
    for (int i = 0; i < 4; ++i) {
      const int e = tid + i * 128;                 // 16B chunk 0..511
      const int r = e >> 3, c = (e & 7) * 8;
      const v8h chunk = *(const v8h*)(lds + r * 64 + c);
      if (p == 0)      *(v8h*)(Qh + ((size_t)bh * S_ + s0 + r) * DH_ + c) = chunk;
      else if (p == 1) *(v8h*)(Kh + ((size_t)bh * S_ + s0 + r) * DH_ + c) = chunk;
      else             *(v8h*)(Vt + ((size_t)bh * DH_ + r) * S_ + s0 + c) = chunk;
    }
    __syncthreads();
  }
}

// ---------------------------------------------------------------------------
// Kernel 2: causal flash attention.  grid = (B*H, S/64), block = 128.
// Each wave owns a 16-query tile; loops over 32-key blocks with an online
// f32 softmax; score & PV GEMMs on WMMA.  No block-level syncs (waves have
// divergent causal trip counts) -- LDS scratch is strictly per-wave.
// ---------------------------------------------------------------------------
__global__ __launch_bounds__(128) void attn_kernel(
    const _Float16* __restrict__ Qh, const _Float16* __restrict__ Kh,
    const _Float16* __restrict__ Vt, _Float16* __restrict__ Zh) {
  __shared__ __align__(32) _Float16 lds2[4][1024];  // per-wave: P stage + Z stage

  const int bh    = blockIdx.x;
  const int tid   = threadIdx.x;
  const int lane  = tid & 31;
  const int w     = tid >> 5;
  const int half_ = lane >> 4;
  const int l16   = lane & 15;
  const int q0    = blockIdx.y * 64 + w * 16;   // this wave's query base

  // Q A-fragments (dh split 0-31 / 32-63), direct contiguous global loads
  const _Float16* qrow = Qh + ((size_t)bh * S_ + q0 + l16) * DH_;
  v16h aq[2];
#pragma unroll
  for (int f = 0; f < 2; ++f)
    aq[f] = mk_frag(qrow + f * 32 + half_ * 16, qrow + f * 32 + half_ * 16 + 8);

  float mrow[8], lrow[8];
#pragma unroll
  for (int r = 0; r < 8; ++r) { mrow[r] = -3.0e38f; lrow[r] = 0.0f; }
  v8f zacc[4] = {};
  const float scale = 0.125f;  // 1/sqrt(64)

  for (int j0 = 0; j0 < q0 + 16; j0 += 32) {
    // ---- scores: two 16-key C tiles, K-dim = DH = 2 WMMAs each
    float sv[2][8];
#pragma unroll
    for (int kt = 0; kt < 2; ++kt) {
      const int key = j0 + kt * 16 + l16;  // this lane's key column (C layout N)
      const _Float16* krow = Kh + ((size_t)bh * S_ + key) * DH_;
      v8f c = {};
#pragma unroll
      for (int f = 0; f < 2; ++f) {
        v16h bk = *(const v16h*)(krow + f * 32 + half_ * 16);
        c = wmma16(aq[f], bk, c);
      }
#pragma unroll
      for (int r = 0; r < 8; ++r) {
        const int q = q0 + half_ * 8 + r;
        sv[kt][r] = (key > q) ? IGNORE_ : c[r] * scale;
      }
    }

    // ---- online softmax (row = M lives across 16 lanes of each half)
    float alpha[8];
#pragma unroll
    for (int r = 0; r < 8; ++r) {
      float lm = fmaxf(sv[0][r], sv[1][r]);
#pragma unroll
      for (int mdist = 1; mdist < 16; mdist <<= 1)
        lm = fmaxf(lm, __shfl_xor(lm, mdist, 32));
      const float mnew = fmaxf(mrow[r], lm);
      const float a    = __expf(mrow[r] - mnew);
      const float p0   = __expf(sv[0][r] - mnew);
      const float p1   = __expf(sv[1][r] - mnew);
      float ls = p0 + p1;
#pragma unroll
      for (int mdist = 1; mdist < 16; mdist <<= 1)
        ls += __shfl_xor(ls, mdist, 32);
      lrow[r]  = lrow[r] * a + ls;
      mrow[r]  = mnew;
      alpha[r] = a;
      sv[0][r] = p0;
      sv[1][r] = p1;
    }
#pragma unroll
    for (int t = 0; t < 4; ++t)
#pragma unroll
      for (int r = 0; r < 8; ++r) zacc[t][r] *= alpha[r];

    // ---- transpose P (C layout) -> A layout via per-wave LDS
#pragma unroll
    for (int kt = 0; kt < 2; ++kt)
#pragma unroll
      for (int r = 0; r < 8; ++r) {
        const int m = half_ * 8 + r;
        lds2[w][m * 32 + kt * 16 + l16] = (_Float16)sv[kt][r];
      }
    __builtin_amdgcn_wave_barrier();  // same-wave DS ops are in-order in HW

    v16h ap = mk_frag(&lds2[w][l16 * 32 + half_ * 16],
                      &lds2[w][l16 * 32 + half_ * 16 + 8]);

    // ---- P(16x32) @ V(32x64): Vt is [bh][dh][s] so B-frags are contiguous
#pragma unroll
    for (int t = 0; t < 4; ++t) {
      const int n = t * 16 + l16;  // dh column
      const _Float16* vrow = Vt + ((size_t)bh * DH_ + n) * S_ + j0 + half_ * 16;
      v16h bv = *(const v16h*)vrow;
      zacc[t] = wmma16(ap, bv, zacc[t]);
    }
  }

  // ---- finalize: /rowsum, per-wave LDS staging, b128 stores of Z [bh][s][dh]
  float inv[8];
#pragma unroll
  for (int r = 0; r < 8; ++r) inv[r] = 1.0f / lrow[r];
#pragma unroll
  for (int t = 0; t < 4; ++t)
#pragma unroll
    for (int r = 0; r < 8; ++r) {
      const int m = half_ * 8 + r;
      lds2[w][m * 64 + t * 16 + l16] = (_Float16)(zacc[t][r] * inv[r]);
    }
  __builtin_amdgcn_wave_barrier();
#pragma unroll
  for (int i = 0; i < 4; ++i) {
    const int e = lane + i * 32;                   // 16B chunk 0..127
    const int r = e >> 3, c = (e & 7) * 8;
    const v8h chunk = *(const v8h*)(&lds2[w][r * 64 + c]);
    *(v8h*)(Zh + ((size_t)bh * S_ + q0 + r) * DH_ + c) = chunk;
  }
}

// ---------------------------------------------------------------------------
// Kernel 3: per-head output projection Z[bh] @ W_O[h] + b_O/H.
// grid = (B*H, S/64, DM/64), block = 128 (4 waves).
// ---------------------------------------------------------------------------
__global__ __launch_bounds__(128) void outproj_kernel(
    const _Float16* __restrict__ Zh, const float* __restrict__ Wo,
    const float* __restrict__ bO, float* __restrict__ out) {
  __shared__ __align__(32) _Float16 ldsw[64 * 64];  // W_O tile transposed [n][k]

  const int bh = blockIdx.x;
  const int b  = bh / H_;
  const int h  = bh % H_;
  const int s0 = blockIdx.y * 64;
  const int c0 = blockIdx.z * 64;

  const int tid   = threadIdx.x;
  const int lane  = tid & 31;
  const int w     = tid >> 5;
  const int half_ = lane >> 4;
  const int l16   = lane & 15;

  // Stage W_O[h][0:64][c0:c0+64] transposed as f16 [n][dh]; float4 loads
#pragma unroll
  for (int i = 0; i < 8; ++i) {
    const int e = tid + i * 128;                   // float4 chunk 0..1023
    const int kr = e >> 4, n = (e & 15) * 4;
    const float4 v = *(const float4*)(Wo + ((size_t)h * DH_ + kr) * DM_ + c0 + n);
    ldsw[(n + 0) * 64 + kr] = (_Float16)v.x;
    ldsw[(n + 1) * 64 + kr] = (_Float16)v.y;
    ldsw[(n + 2) * 64 + kr] = (_Float16)v.z;
    ldsw[(n + 3) * 64 + kr] = (_Float16)v.w;
  }
  __syncthreads();

  const _Float16* zrow = Zh + ((size_t)bh * S_ + s0 + w * 16 + l16) * DH_;
  v16h az[2];
#pragma unroll
  for (int f = 0; f < 2; ++f)
    az[f] = mk_frag(zrow + f * 32 + half_ * 16, zrow + f * 32 + half_ * 16 + 8);

  v8f acc[4] = {};
#pragma unroll
  for (int f = 0; f < 2; ++f)
#pragma unroll
    for (int t = 0; t < 4; ++t) {
      const int n = t * 16 + l16;
      v16h bw = *(const v16h*)(ldsw + n * 64 + f * 32 + half_ * 16);
      acc[t] = wmma16(az[f], bw, acc[t]);
    }

  const float invH = 1.0f / (float)H_;
#pragma unroll
  for (int t = 0; t < 4; ++t) {
    const int n = c0 + t * 16 + l16;
    const float bb = bO[n] * invH;
#pragma unroll
    for (int r = 0; r < 8; ++r) {
      const int s = s0 + w * 16 + half_ * 8 + r;
      // lanes 0-15 write 16 consecutive f32 -> coalesced 64B segments
      out[((size_t)(b * S_ + s) * H_ + h) * DM_ + n] = acc[t][r] + bb;
    }
  }
}

// ---------------------------------------------------------------------------
extern "C" void kernel_launch(void* const* d_in, const int* in_sizes, int n_in,
                              void* d_out, int out_size, void* d_ws,
                              size_t ws_size, hipStream_t stream) {
  const float* x  = (const float*)d_in[0];
  const float* Wq = (const float*)d_in[1];
  const float* bq = (const float*)d_in[2];
  const float* Wk = (const float*)d_in[3];
  const float* bk = (const float*)d_in[4];
  const float* Wv = (const float*)d_in[5];
  const float* bv = (const float*)d_in[6];
  const float* Wo = (const float*)d_in[7];
  const float* bO = (const float*)d_in[8];
  float* out = (float*)d_out;

  const size_t NQKV = (size_t)B_ * H_ * S_ * DH_;  // 3,145,728 f16 each
  _Float16* Qh = (_Float16*)d_ws;
  _Float16* Kh = Qh + NQKV;
  _Float16* Vt = Kh + NQKV;
  _Float16* Zh = Vt + NQKV;   // total 25.2 MB scratch, L2-resident (192 MB L2)

  dim3 g1(B_ * H_, S_ / 64);
  qkv_kernel<<<g1, 128, 0, stream>>>(x, Wq, bq, Wk, bk, Wv, bv, Qh, Kh, Vt);

  dim3 g2(B_ * H_, S_ / 64);
  attn_kernel<<<g2, 128, 0, stream>>>(Qh, Kh, Vt, Zh);

  dim3 g3(B_ * H_, S_ / 64, DM_ / 64);
  outproj_kernel<<<g3, 128, 0, stream>>>(Zh, Wo, bO, out);
}
